// GatedMLP_37907381355066
// MI455X (gfx1250) — matrix operand order repaired and checked by reading, not compiled, not run on hardware
//
#include <hip/hip_runtime.h>

typedef unsigned short u16;
typedef unsigned int   u32;

typedef __attribute__((ext_vector_type(16))) __bf16 v16bf;
typedef __attribute__((ext_vector_type(8)))  float  v8f;
typedef __attribute__((ext_vector_type(4)))  float  v4f;
typedef __attribute__((ext_vector_type(4)))  int    v4i;
typedef __attribute__((ext_vector_type(4)))  unsigned int v4u;
typedef __attribute__((ext_vector_type(2)))  unsigned int v2u;

#define HIDDEN 4096
#define INTERM 14336
#define NTOK   4096   /* B*S = 2*2048 */
#define BM 128
#define BN 64
#define BK 32
#define LDT 40        /* padded LDS row length in bf16 elems (80 B, 16B-aligned, bank-decorrelated) */

// ---------- fp32 -> bf16 (round to nearest even) ----------
static __device__ __forceinline__ u32 pk_bf16(float a, float b) {
  u32 ua = __float_as_uint(a), ub = __float_as_uint(b);
  ua += 0x7fffu + ((ua >> 16) & 1u);
  ub += 0x7fffu + ((ub >> 16) & 1u);
  return (ua >> 16) | (ub & 0xffff0000u);
}
static __device__ __forceinline__ u16 f2bf(float x) {
  u32 u = __float_as_uint(x);
  u += 0x7fffu + ((u >> 16) & 1u);
  return (u16)(u >> 16);
}

// ---------- async global->LDS (CDNA5), guarded with sync fallback ----------
#if __has_builtin(__builtin_amdgcn_global_load_async_to_lds_b128)
#define HAS_ASYNC_LDS 1
#else
#define HAS_ASYNC_LDS 0
#endif

static __device__ __forceinline__ void g2l_b128(const u16* g, u16* l) {
#if HAS_ASYNC_LDS
  __builtin_amdgcn_global_load_async_to_lds_b128(
      (__attribute__((address_space(1))) v4i*)g,
      (__attribute__((address_space(3))) v4i*)l, 0, 0);
#else
  *(v4u*)l = *(const v4u*)g;
#endif
}

template <int N>
static __device__ __forceinline__ void wait_async() {
#if HAS_ASYNC_LDS
#if __has_builtin(__builtin_amdgcn_s_wait_asynccnt)
  __builtin_amdgcn_s_wait_asynccnt(N);
#else
  asm volatile("s_wait_asynccnt %0" ::"i"(N) : "memory");
#endif
#endif
}

union Frag {
  v16bf v;
  v4u   q[2];
};

// ---------- fp32 -> bf16 bulk conversion ----------
__global__ __launch_bounds__(256) void k_cvt(const v4f* __restrict__ src,
                                             v2u* __restrict__ dst, int n4) {
  int i = blockIdx.x * blockDim.x + threadIdx.x;
  int stride = gridDim.x * blockDim.x;
  for (; i < n4; i += stride) {
    v4f v = src[i];
    v2u r;
    r.x = pk_bf16(v.x, v.y);
    r.y = pk_bf16(v.z, v.w);
    dst[i] = r;
  }
}

// ---------- WMMA GEMM:  C[M,Ntot-slice] = A[M,K] * B[N,K]^T  (both K-contiguous) ----------
// NMAT==2: B0=gate, B1=up; epilogue a = silu(g)*u stored bf16.
// NMAT==1: plain GEMM; epilogue stores fp32.
template <int NMAT>
__global__ __launch_bounds__(256) void gemm_bf16(const u16* __restrict__ A,
                                                 const u16* __restrict__ B0,
                                                 const u16* __restrict__ B1,
                                                 void* __restrict__ outp,
                                                 int K, int Ntot) {
  __shared__ u16 As[2][BM][LDT];
  __shared__ u16 Bs[2][NMAT][BN][LDT];

  const int tid   = threadIdx.x;
  const int lane  = tid & 31;
  const int wid   = tid >> 5;
  const int waveM = wid & 3;   // 4 slabs of 32 rows
  const int waveN = wid >> 2;  // 2 slabs of 32 cols
  const int lr    = lane & 15;
  const int hi    = lane >> 4; // 0: lanes 0-15, 1: lanes 16-31

  const int tileM = blockIdx.x * BM;
  const int tileN = blockIdx.y * BN;

  // ---- branch-free cooperative copy: each thread owns fixed (row, chunk) coords ----
  const int prow = tid >> 2;        // 0..63
  const int pcc  = (tid & 3) * 8;   // bf16 element offset of this thread's 16B chunk
  const u16* gA0 = &A[(size_t)(tileM + prow) * K + pcc];
  const u16* gA1 = &A[(size_t)(tileM + prow + 64) * K + pcc];
  const u16* gB0 = &B0[(size_t)(tileN + prow) * K + pcc];
  const u16* gB1 = (NMAT == 2) ? &B1[(size_t)(tileN + prow) * K + pcc] : gB0;

  auto prefetch = [&](int buf, int kk) {
    const int k0 = kk * BK;
    g2l_b128(gA0 + k0, &As[buf][prow][pcc]);
    g2l_b128(gA1 + k0, &As[buf][prow + 64][pcc]);
    g2l_b128(gB0 + k0, &Bs[buf][0][prow][pcc]);
    if constexpr (NMAT == 2) g2l_b128(gB1 + k0, &Bs[buf][NMAT - 1][prow][pcc]);
  };

  const v8f vzero = {0.f, 0.f, 0.f, 0.f, 0.f, 0.f, 0.f, 0.f};
  v8f acc[NMAT][2][2];
  #pragma unroll
  for (int m = 0; m < NMAT; ++m)
    #pragma unroll
    for (int a = 0; a < 2; ++a)
      #pragma unroll
      for (int b = 0; b < 2; ++b) acc[m][a][b] = vzero;

  const int nk = K / BK;
  prefetch(0, 0);

  for (int kk = 0; kk < nk; ++kk) {
    const int buf = kk & 1;
    const bool more = (kk + 1 < nk);
    if (more) {
      prefetch(buf ^ 1, kk + 1);       // overlap next-stage DMA with this stage
      wait_async<2 + NMAT>();          // current stage's copies (issued first) done
    } else {
      wait_async<0>();
    }
    __syncthreads();                   // all waves' LDS writes visible

    // A fragments (16x32 bf16): lanes 0-15 -> K 0-7 & 16-23; lanes 16-31 -> K 8-15 & 24-31
    Frag af[2];
    #pragma unroll
    for (int mt = 0; mt < 2; ++mt) {
      const u16* p = &As[buf][waveM * 32 + mt * 16 + lr][hi * 8];
      af[mt].q[0] = *(const v4u*)p;
      af[mt].q[1] = *(const v4u*)(p + 16);
    }
    // B fragments (32x16 bf16): lanes 0-15 -> K 0-7 & 16-23; lanes 16-31 -> K 8-15 & 24-31
    Frag bfr[NMAT][2];
    #pragma unroll
    for (int m = 0; m < NMAT; ++m)
      #pragma unroll
      for (int nt = 0; nt < 2; ++nt) {
        const u16* p = &Bs[buf][m][waveN * 32 + nt * 16 + lr][hi * 8];
        bfr[m][nt].q[0] = *(const v4u*)p;
        bfr[m][nt].q[1] = *(const v4u*)(p + 16);
      }

    #pragma unroll
    for (int m = 0; m < NMAT; ++m)
      #pragma unroll
      for (int mt = 0; mt < 2; ++mt)
        #pragma unroll
        for (int nt = 0; nt < 2; ++nt)
          acc[m][mt][nt] = __builtin_amdgcn_wmma_f32_16x16x32_bf16(
              false, af[mt].v, false, bfr[m][nt].v, (short)0,
              acc[m][mt][nt], false, false);

    __syncthreads();                   // done reading buf before it is overwritten
  }

  // ---- epilogue ----
  // C/D layout: element i of v8f is row (i + 8*hi), col = lr within the 16x16 tile.
  if constexpr (NMAT == 2) {
    u16* aout = (u16*)outp;
    #pragma unroll
    for (int mt = 0; mt < 2; ++mt)
      #pragma unroll
      for (int nt = 0; nt < 2; ++nt) {
        v8f g = acc[0][mt][nt];
        v8f u = acc[1][mt][nt];
        int rb  = tileM + waveM * 32 + mt * 16 + hi * 8;
        int col = tileN + waveN * 32 + nt * 16 + lr;
        #pragma unroll
        for (int i = 0; i < 8; ++i) {
          float gv = g[i], uv = u[i];
          float s = gv / (1.f + __expf(-gv));  // silu
          aout[(size_t)(rb + i) * Ntot + col] = f2bf(s * uv);
        }
      }
  } else {
    float* dout = (float*)outp;
    #pragma unroll
    for (int mt = 0; mt < 2; ++mt)
      #pragma unroll
      for (int nt = 0; nt < 2; ++nt) {
        v8f c = acc[0][mt][nt];
        int rb  = tileM + waveM * 32 + mt * 16 + hi * 8;
        int col = tileN + waveN * 32 + nt * 16 + lr;
        #pragma unroll
        for (int i = 0; i < 8; ++i)
          dout[(size_t)(rb + i) * Ntot + col] = c[i];
      }
  }
}

extern "C" void kernel_launch(void* const* d_in, const int* in_sizes, int n_in,
                              void* d_out, int out_size, void* d_ws, size_t ws_size,
                              hipStream_t stream) {
  (void)in_sizes; (void)n_in; (void)out_size; (void)ws_size;
  const float* x  = (const float*)d_in[0];
  const float* Wg = (const float*)d_in[1];
  const float* Wu = (const float*)d_in[2];
  const float* Wd = (const float*)d_in[3];

  const size_t xN = (size_t)NTOK * HIDDEN;    // 16.8M elems
  const size_t wN = (size_t)INTERM * HIDDEN;  // 58.7M elems
  const size_t aN = (size_t)NTOK * INTERM;    // 58.7M elems

  char* ws = (char*)d_ws;
  u16* xb  = (u16*)ws; ws += xN * sizeof(u16);
  u16* Wgb = (u16*)ws; ws += wN * sizeof(u16);
  u16* Wub = (u16*)ws; ws += wN * sizeof(u16);
  u16* Wdb = (u16*)ws; ws += wN * sizeof(u16);
  u16* ab  = (u16*)ws; ws += aN * sizeof(u16);
  // total ws use: ~480 MB

  // fp32 -> bf16 pre-pass (memory-bound, ~40us; compute stages are ~300us)
  k_cvt<<<2048, 256, 0, stream>>>((const v4f*)x,  (v2u*)xb,  (int)(xN / 4));
  k_cvt<<<4096, 256, 0, stream>>>((const v4f*)Wg, (v2u*)Wgb, (int)(wN / 4));
  k_cvt<<<4096, 256, 0, stream>>>((const v4f*)Wu, (v2u*)Wub, (int)(wN / 4));
  k_cvt<<<4096, 256, 0, stream>>>((const v4f*)Wd, (v2u*)Wdb, (int)(wN / 4));

  // a = silu(x Wg^T) * (x Wu^T)   [4096 x 14336] bf16
  gemm_bf16<2><<<dim3(NTOK / BM, INTERM / BN), 256, 0, stream>>>(
      xb, Wgb, Wub, (void*)ab, HIDDEN, INTERM);

  // d = a Wd^T                     [4096 x 4096] fp32
  gemm_bf16<1><<<dim3(NTOK / BM, HIDDEN / BN), 256, 0, stream>>>(
      ab, Wdb, nullptr, d_out, INTERM, HIDDEN);
}
